// SPINN_46359876993764
// MI455X (gfx1250) — matrix operand order repaired and checked by reading, not compiled, not run on hardware
//
#include <hip/hip_runtime.h>
#include <hip/hip_bf16.h>
#include <stddef.h>

// ---------------------------------------------------------------------------
// SPINN on MI455X (gfx1250, wave32).
// Fixed left-branching parse => recurrence: acc = TreeLSTM(acc, word_k).
//   K1: proj = sentence @ W_word + b_word      (WMMA bf16/f32-acc)
//       -> h_buf word-major bf16 [l][b][256], c_buf tile-packed f32 fragments
//   K2: R = h_buf @ W_right + b_reduce         (WMMA, tiled over (k, b-tile))
//       -> stored in WMMA D-fragment order so the recurrence seeds its
//          accumulators with two contiguous b128 loads per gate
//   K3: 511-step serial chain, one 1024-thread workgroup (32 waves, 8/SIMD)
//       on one WGP. h double-buffered in LDS (1 barrier/step), cell state in
//       registers, W_left bf16 tiles streamed from L2 (640 KB resident).
// ---------------------------------------------------------------------------

#define BATCH 64
#define SEQL  512
#define EMB   300
#define EMBP  320   // padded to 10 K-tiles of 32
#define HID   256
#define GATE5 1280  // 5*HID
#define ROWS  (BATCH * SEQL)   // 32768
#define HSLAB (BATCH * HID)    // 16384 elements per word slab

typedef __attribute__((ext_vector_type(16))) __bf16 v16bf;
typedef __attribute__((ext_vector_type(8)))  float  v8f;

union Frag16 { int4 q[2]; v16bf v; };   // 16 x bf16 WMMA A/B operand
union Frag32 { int4 q[2]; v8f   v; };   // 8 x f32 WMMA C/D fragment

__device__ __forceinline__ unsigned short f2bf(float x) {
    unsigned int u = __float_as_uint(x);
    unsigned int r = u + 0x7FFFu + ((u >> 16) & 1u);   // round-to-nearest-even
    return (unsigned short)(r >> 16);
}
__device__ __forceinline__ float sigm(float x) { return 1.0f / (1.0f + __expf(-x)); }

// --------------------------- packing kernels -------------------------------

// sentence [ROWS,300] f32 -> bf16 padded [ROWS,320]
__global__ void spinn_pack_sentence(const float* __restrict__ sent,
                                    unsigned short* __restrict__ out) {
    size_t i = (size_t)blockIdx.x * blockDim.x + threadIdx.x;
    if (i >= (size_t)ROWS * EMBP) return;
    int row = (int)(i / EMBP);
    int kp  = (int)(i % EMBP);
    float v = (kp < EMB) ? sent[(size_t)row * EMB + kp] : 0.0f;
    out[i] = f2bf(v);
}

// W [Kreal,N] f32 -> WMMA-B packed tiles [KT][NT][lane32][slot16] bf16.
// B operand (32x16 bf16): lane half h, slot s -> K = 8h+s (s<8), 16+8h+(s-8) (s>=8);
// column = lane&15.
__global__ void spinn_pack_w(const float* __restrict__ W,
                             unsigned short* __restrict__ out,
                             int Kreal, int N, int KT, int NT) {
    size_t i = (size_t)blockIdx.x * blockDim.x + threadIdx.x;
    if (i >= (size_t)KT * NT * 512) return;
    int tile = (int)(i >> 9);
    int sl   = (int)(i & 511);
    int lane = sl >> 4;
    int slot = sl & 15;
    int kt = tile / NT;
    int nt = tile - kt * NT;
    int k  = kt * 32 + (slot >> 3) * 16 + (lane >> 4) * 8 + (slot & 7);
    int n  = nt * 16 + (lane & 15);
    float v = (k < Kreal) ? W[(size_t)k * N + n] : 0.0f;
    out[i] = f2bf(v);
}

// --------------------------- projection GEMM -------------------------------
// proj = sentence @ W_word + b_word.  M=32768 (rows b*512+l), N=512, K=320.
// Epilogue: n<256 -> h to word-major bf16 hbuf_t[(l*64+b)*256+n];
//           n>=256 -> c to tile-packed fragment order
//           c_pk[((l*4 + b/16)*16 + j/16)*256 + lane*8 + r].
__global__ void spinn_proj_wmma(const unsigned short* __restrict__ A,
                                const unsigned short* __restrict__ Bpk,
                                const float* __restrict__ bias,
                                unsigned short* __restrict__ hbuf_t,
                                float* __restrict__ c_pk) {
    const int lane = threadIdx.x & 31;
    const int wid  = (int)((blockIdx.x * blockDim.x + threadIdx.x) >> 5);
    const int rowTile = wid >> 3;          // 2048 row tiles
    const int ng      = wid & 7;           // 8 groups of 64 cols
    const int rowbase = rowTile << 4;
    const int nt0     = ng << 2;
    const int mrow = rowbase + (lane & 15);
    const int kq   = lane >> 4;

    v8f acc[4] = {};
    for (int kt = 0; kt < 10; ++kt) {
        Frag16 af;
        const int4* ap = reinterpret_cast<const int4*>(
            A + (size_t)mrow * EMBP + kt * 32 + kq * 8);
        af.q[0] = ap[0];
        af.q[1] = ap[2];
#pragma unroll
        for (int t = 0; t < 4; ++t) {
            Frag16 bw;
            const int4* bp = reinterpret_cast<const int4*>(
                Bpk + ((size_t)(kt * 32 + nt0 + t) << 9) + lane * 16);
            bw.q[0] = bp[0];
            bw.q[1] = bp[1];
            acc[t] = __builtin_amdgcn_wmma_f32_16x16x32_bf16(
                false, af.v, false, bw.v, (short)0, acc[t], false, false);
        }
    }

#pragma unroll
    for (int t = 0; t < 4; ++t) {
#pragma unroll
        for (int r = 0; r < 8; ++r) {
            int m = rowbase + r + ((lane >> 4) << 3);   // row = b*512 + l
            int b = m >> 9, l = m & 511;
            int n = (nt0 + t) * 16 + (lane & 15);
            float v = acc[t][r] + bias[n];
            if (n < HID) {
                hbuf_t[((size_t)l * BATCH + b) * HID + n] = f2bf(v);
            } else {
                int j = n - HID;
                int slot_lane = (j & 15) + (((b >> 3) & 1) << 4);
                size_t idx = (((size_t)l * 4 + (b >> 4)) * 16 + (j >> 4)) * 256
                             + slot_lane * 8 + (b & 7);
                c_pk[idx] = v;
            }
        }
    }
}

// ------------------------------ R GEMM -------------------------------------
// R = h_buf @ W_right + b_reduce, tiled over (word k, batch m-tile, 64-col
// group). Output stored in WMMA D-fragment order:
//   R_pk[((k*4+mtile)*80 + ntile)*256 + lane*8 + r]
__global__ void spinn_rgemm_wmma(const unsigned short* __restrict__ hbuf_t,
                                 const unsigned short* __restrict__ Bpk,
                                 const float* __restrict__ bias,
                                 float* __restrict__ R_pk) {
    const int lane = threadIdx.x & 31;
    const int wid  = (int)((blockIdx.x * blockDim.x + threadIdx.x) >> 5);
    const int ng    = wid % 20;            // 20 groups of 64 cols (N=1280)
    const int rest  = wid / 20;
    const int mtile = rest & 3;            // 4 batch tiles of 16
    const int k     = rest >> 2;           // word index 0..511
    const int nt0   = ng << 2;
    const int mrow = (mtile << 4) + (lane & 15);
    const int kq   = lane >> 4;

    const unsigned short* Abase = hbuf_t + ((size_t)k * BATCH + mrow) * HID;

    v8f acc[4] = {};
    for (int kt = 0; kt < 8; ++kt) {
        Frag16 af;
        const int4* ap = reinterpret_cast<const int4*>(Abase + kt * 32 + kq * 8);
        af.q[0] = ap[0];
        af.q[1] = ap[2];
#pragma unroll
        for (int t = 0; t < 4; ++t) {
            Frag16 bw;
            const int4* bp = reinterpret_cast<const int4*>(
                Bpk + ((size_t)(kt * 80 + nt0 + t) << 9) + lane * 16);
            bw.q[0] = bp[0];
            bw.q[1] = bp[1];
            acc[t] = __builtin_amdgcn_wmma_f32_16x16x32_bf16(
                false, af.v, false, bw.v, (short)0, acc[t], false, false);
        }
    }

#pragma unroll
    for (int t = 0; t < 4; ++t) {
        const int nt = nt0 + t;
        const float bv = bias[nt * 16 + (lane & 15)];   // column bias, same all r
        Frag32 s;
#pragma unroll
        for (int r = 0; r < 8; ++r) s.v[r] = acc[t][r] + bv;
        int4* rp = reinterpret_cast<int4*>(
            R_pk + (((size_t)k * 4 + mtile) * 80 + nt) * 256 + lane * 8);
        rp[0] = s.q[0];
        rp[1] = s.q[1];
    }
}

// --------------------------- serial recurrence -----------------------------
// One 1024-thread block (32 waves, 8/SIMD). 64 tile-tasks/step: task t ->
// mtile=t>>4, nt=t&15; wave w owns tasks {2w,2w+1} (same mtile). Per task the
// 5 gate accumulators are independent -> back-to-back WMMAs per A-fragment.
// h double-buffered in LDS (one barrier/step); cell state lives in registers.
__global__ void __launch_bounds__(1024, 1)
spinn_recur_wmma(const unsigned short* __restrict__ hbuf_t,
                 const unsigned short* __restrict__ Wl_pk,
                 const float* __restrict__ R_pk,
                 const float* __restrict__ c_pk,
                 float* __restrict__ out) {
    extern __shared__ char smem[];
    unsigned short* sh = (unsigned short*)smem;      // 2 x [64][256] bf16

    const int tid  = threadIdx.x;
    const int lane = tid & 31;
    const int wid  = tid >> 5;

    const int task0 = wid * 2;
    const int mtile = task0 >> 4;          // same for both tasks of this wave
    const int ntA   = task0 & 15;
    const int mrow  = (mtile << 4) + (lane & 15);
    const int kq    = lane >> 4;
    const int dlocal = ((lane >> 4) << 3);
    const int ncol   = lane & 15;

    // init: word-0 h into LDS buffer 0 (k=0 slab is the first 16384 elems)
    for (int i = tid; i < HSLAB; i += 1024) sh[i] = hbuf_t[i];

    // init: word-0 cell state into registers (tile-packed fragment order)
    float creg[16];
#pragma unroll
    for (int t = 0; t < 2; ++t) {
        Frag32 cf;
        const int4* cp = reinterpret_cast<const int4*>(
            c_pk + ((size_t)mtile * 16 + (ntA + t)) * 256 + lane * 8);
        cf.q[0] = cp[0];
        cf.q[1] = cp[1];
#pragma unroll
        for (int r = 0; r < 8; ++r) creg[t * 8 + r] = cf.v[r];
    }

    for (int k = 1; k < SEQL; ++k) {
        __syncthreads();                              // prev-step h visible
        const unsigned short* shr = sh + ((size_t)((k + 1) & 1)) * HSLAB; // read
        unsigned short*       shw = sh + ((size_t)(k & 1)) * HSLAB;       // write

#pragma unroll
        for (int t = 0; t < 2; ++t) {
            const int nt = ntA + t;
            const size_t seed0 =
                (((size_t)k * 4 + mtile) * 80 + nt) * 256 + lane * 8;
            const size_t cidx =
                (((size_t)k * 4 + mtile) * 16 + nt) * 256 + lane * 8;

            // warm next step's seed/cell tiles (global_prefetch_b8)
            if (k + 1 < SEQL) {
                __builtin_prefetch(R_pk + seed0 + (size_t)4 * 80 * 256, 0, 1);
                __builtin_prefetch(c_pk + cidx + (size_t)4 * 16 * 256, 0, 1);
            }

            // seed gate accumulators with R fragments (2 x b128 per gate)
            v8f acc[5];
#pragma unroll
            for (int g = 0; g < 5; ++g) {
                Frag32 ra;
                const int4* rp = reinterpret_cast<const int4*>(
                    R_pk + seed0 + (size_t)g * 16 * 256);
                ra.q[0] = rp[0];
                ra.q[1] = rp[1];
                acc[g] = ra.v;
            }
            // word-k cell fragment (2 x b128)
            Frag32 crf;
            {
                const int4* cp = reinterpret_cast<const int4*>(c_pk + cidx);
                crf.q[0] = cp[0];
                crf.q[1] = cp[1];
            }

            // g += h_acc @ W_left : per A-fragment, 5 independent WMMAs
#pragma unroll
            for (int kt = 0; kt < 8; ++kt) {
                Frag16 af;
                const int4* ap = reinterpret_cast<const int4*>(
                    shr + mrow * HID + kt * 32 + kq * 8);
                af.q[0] = ap[0];
                af.q[1] = ap[2];
#pragma unroll
                for (int g = 0; g < 5; ++g) {
                    Frag16 bw;
                    const int4* bp = reinterpret_cast<const int4*>(
                        Wl_pk + ((size_t)(kt * 80 + g * 16 + nt) << 9) + lane * 16);
                    bw.q[0] = bp[0];
                    bw.q[1] = bp[1];
                    acc[g] = __builtin_amdgcn_wmma_f32_16x16x32_bf16(
                        false, af.v, false, bw.v, (short)0, acc[g], false, false);
                }
            }

            // TreeLSTM gates (f32)
#pragma unroll
            for (int r = 0; r < 8; ++r) {
                int b = (mtile << 4) + r + dlocal;
                int j = nt * 16 + ncol;
                float ig = acc[0][r], fl = acc[1][r], fr = acc[2][r];
                float og = acc[3][r], ug = acc[4][r];
                float cl = creg[t * 8 + r];
                float cr = crf.v[r];
                float cn = sigm(fl) * cl + sigm(fr) * cr + sigm(ig) * tanhf(ug);
                float hn = sigm(og) * tanhf(cn);
                creg[t * 8 + r] = cn;
                shw[b * HID + j] = f2bf(hn);
                if (k == SEQL - 1) out[b * HID + j] = hn;
            }
        }
    }
}

// ------------------------------- launcher ----------------------------------

extern "C" void kernel_launch(void* const* d_in, const int* in_sizes, int n_in,
                              void* d_out, int out_size, void* d_ws, size_t ws_size,
                              hipStream_t stream) {
    (void)in_sizes; (void)n_in; (void)out_size; (void)ws_size;
    const float* sentence = (const float*)d_in[0];
    // d_in[1] = transitions: fixed left-branching pattern, not needed
    const float* W_word   = (const float*)d_in[2];
    const float* b_word   = (const float*)d_in[3];
    const float* W_left   = (const float*)d_in[4];
    const float* W_right  = (const float*)d_in[5];
    const float* b_reduce = (const float*)d_in[6];
    float* out = (float*)d_out;

    char* ws = (char*)d_ws;
    size_t off = 0;
    unsigned short* sentA    = (unsigned short*)(ws + off); off += (size_t)ROWS * EMBP * 2;    // 20.97 MB
    unsigned short* Wword_pk = (unsigned short*)(ws + off); off += (size_t)10 * 32 * 512 * 2;  // 0.33 MB
    unsigned short* Wleft_pk = (unsigned short*)(ws + off); off += (size_t)8 * 80 * 512 * 2;   // 0.66 MB
    unsigned short* Wright_pk= (unsigned short*)(ws + off); off += (size_t)8 * 80 * 512 * 2;   // 0.66 MB
    unsigned short* hbuf_t   = (unsigned short*)(ws + off); off += (size_t)ROWS * HID * 2;     // 16.8 MB
    float*          c_pk     = (float*)(ws + off);          off += (size_t)ROWS * HID * 4;     // 33.6 MB
    float*          R_pk     = (float*)(ws + off);          off += (size_t)ROWS * GATE5 * 4;   // 167.8 MB

    // 1) pack / convert
    {
        size_t total = (size_t)ROWS * EMBP;
        spinn_pack_sentence<<<dim3((unsigned)((total + 255) / 256)), dim3(256), 0, stream>>>(
            sentence, sentA);
    }
    {
        size_t total = (size_t)10 * 32 * 512;
        spinn_pack_w<<<dim3((unsigned)((total + 255) / 256)), dim3(256), 0, stream>>>(
            W_word, Wword_pk, EMB, 2 * HID, 10, 32);
    }
    {
        size_t total = (size_t)8 * 80 * 512;
        spinn_pack_w<<<dim3((unsigned)((total + 255) / 256)), dim3(256), 0, stream>>>(
            W_left, Wleft_pk, HID, GATE5, 8, 80);
        spinn_pack_w<<<dim3((unsigned)((total + 255) / 256)), dim3(256), 0, stream>>>(
            W_right, Wright_pk, HID, GATE5, 8, 80);
    }

    // 2) proj (M=32768, K=320, N=512): 16384 waves
    spinn_proj_wmma<<<dim3(2048), dim3(256), 0, stream>>>(
        sentA, Wword_pk, b_word, hbuf_t, c_pk);

    // 3) R GEMM (512 words x 4 m-tiles x 20 n-groups): 40960 waves
    spinn_rgemm_wmma<<<dim3(5120), dim3(256), 0, stream>>>(
        hbuf_t, Wright_pk, b_reduce, R_pk);

    // 4) serial TreeLSTM chain, one workgroup, 64 KB LDS (h double-buffer)
    {
        size_t lds = (size_t)2 * HSLAB * sizeof(unsigned short);   // 65536
        spinn_recur_wmma<<<dim3(1), dim3(1024), lds, stream>>>(
            hbuf_t, Wleft_pk, R_pk, c_pk, out);
    }
}
